// VariableSelectionNetwork_82360292868210
// MI455X (gfx1250) — matrix operand order repaired and checked by reading, not compiled, not run on hardware
//
#include <hip/hip_runtime.h>
#include <hip/hip_bf16.h>

typedef __attribute__((ext_vector_type(16))) __bf16   v16bf;
typedef __attribute__((ext_vector_type(8)))  float    v8f;
typedef __attribute__((ext_vector_type(8)))  unsigned v8u;

#define F_DIM 32
#define U_DIM 256
#define NTOK  8192          // B*T = 64*128
#define TOKS_PER_WG 16
#define NBLOB (32*8*16)     // f * kk * nn  = 4096 B-tiles
#define WS_HI_DWORDS ((size_t)NBLOB * 256)

__device__ __forceinline__ unsigned fasu(float x) { return __builtin_bit_cast(unsigned, x); }
__device__ __forceinline__ float    uasf(unsigned x) { return __builtin_bit_cast(float, x); }

// round-to-nearest-even f32 -> bf16 bits
__device__ __forceinline__ unsigned bf16_rne(float x) {
    unsigned u = fasu(x);
    return (u + 0x7FFFu + ((u >> 16) & 1u)) >> 16;
}

__device__ __forceinline__ float sigmoidf(float x) {
    return 1.0f / (1.0f + __expf(-x));
}

// ---------------------------------------------------------------------------
// Prep: split W2[f][k][n] (f32) into bf16 hi/lo, laid out in this kernel's contiguous-K B-tile
// VGPR map: blob=(f*8+kk)*16+nn ; dword (lane*8 + j) holds K=(lane>>4)*16+2j
// (low half) and K+1 (high half), N = nn*16 + (lane&15), K-base = kk*32.
// Each lane of the main kernel then loads its 8 dwords with one 32B read.
// ---------------------------------------------------------------------------
__global__ __launch_bounds__(256) void vsn_prep_w2(const float* __restrict__ W2,
                                                   unsigned* __restrict__ bhi,
                                                   unsigned* __restrict__ blo) {
    const int blob = blockIdx.x;          // 0..4095
    const int idx  = threadIdx.x;         // 0..255  == lane*8 + j
    const int lane = idx >> 3;
    const int j    = idx & 7;
    const int f  = blob >> 7;
    const int kk = (blob >> 4) & 7;
    const int nn = blob & 15;

    const int n  = nn * 16 + (lane & 15);
    const int k0 = kk * 32 + ((lane >> 4) * 16) + 2 * j;

    const float* base = W2 + (size_t)f * (U_DIM * U_DIM);
    float v0 = base[(size_t)k0 * U_DIM + n];
    float v1 = base[(size_t)(k0 + 1) * U_DIM + n];

    // hi = truncated-to-bf16 (exact in f32), lo = RNE(residual)
    unsigned h0 = fasu(v0) & 0xFFFF0000u;
    unsigned h1 = fasu(v1) & 0xFFFF0000u;
    float r0 = v0 - uasf(h0);
    float r1 = v1 - uasf(h1);

    size_t off = (size_t)blob * 256 + idx;
    bhi[off] = (h1) | (h0 >> 16);                         // [odd K | even K]
    blo[off] = (bf16_rne(r1) << 16) | bf16_rne(r0);
}

// ---------------------------------------------------------------------------
// Main fused kernel: one WG = 16 tokens, 16 waves, 2 features per wave.
// ---------------------------------------------------------------------------
__global__ __launch_bounds__(512) void vsn_main(
    const float* __restrict__ x,   const float* __restrict__ W1,
    const float* __restrict__ b1,  const float* __restrict__ b2,
    const float* __restrict__ Wg,  const float* __restrict__ bg,
    const float* __restrict__ gma, const float* __restrict__ bta,
    const float* __restrict__ Ws,  const float* __restrict__ bs,
    const unsigned* __restrict__ bhi, const unsigned* __restrict__ blo,
    float* __restrict__ out) {

    extern __shared__ char smem[];
    _Float16* lnbuf = (_Float16*)smem;                          // [16][32][256] fp16
    float*    Mbuf  = (float*)(smem + TOKS_PER_WG * F_DIM * U_DIM * 2);  // [16][32]
    float*    wbuf  = Mbuf + TOKS_PER_WG * F_DIM;                        // [16][32]

    const int tid  = threadIdx.x;
    const int lane = tid & 31;
    const int wave = tid >> 5;
    const int half = lane >> 4;       // A/B/C half-lane group
    const int l15  = lane & 15;
    const int tok0 = blockIdx.x * TOKS_PER_WG;

#pragma unroll 1
    for (int fi = 0; fi < 2; ++fi) {
        const int f = wave + fi * 16;

        // x for A layout (M = lane&15) and for C layout (M = j + 8*half)
        const float xA = x[(size_t)(tok0 + l15) * F_DIM + f];
        float xv[8];
#pragma unroll
        for (int j = 0; j < 8; ++j)
            xv[j] = x[(size_t)(tok0 + j + 8 * half) * F_DIM + f];

        v8f acc[16];
#pragma unroll
        for (int nn = 0; nn < 16; ++nn)
            acc[nn] = (v8f){0.f, 0.f, 0.f, 0.f, 0.f, 0.f, 0.f, 0.f};

        const float* W1f = W1 + f * U_DIM;
        const float* b1f = b1 + f * U_DIM;
        const unsigned* bhiF = bhi + (size_t)(f * 8) * 16 * 256;
        const unsigned* bloF = blo + (size_t)(f * 8) * 16 * 256;

        for (int kk = 0; kk < 8; ++kk) {
            // Build A tile (16x32) in contiguous-K order: VGPR j holds K = base,base+1
            unsigned ah[8], al[8];
#pragma unroll
            for (int j = 0; j < 8; ++j) {
                int k0 = kk * 32 + half * 16 + j * 2;
                float a0 = fmaxf(fmaf(xA, W1f[k0],     b1f[k0]),     0.f);
                float a1 = fmaxf(fmaf(xA, W1f[k0 + 1], b1f[k0 + 1]), 0.f);
                unsigned h0 = fasu(a0) & 0xFFFF0000u;
                unsigned h1 = fasu(a1) & 0xFFFF0000u;
                float r0 = a0 - uasf(h0);
                float r1 = a1 - uasf(h1);
                ah[j] = h1 | (h0 >> 16);
                al[j] = (bf16_rne(r1) << 16) | bf16_rne(r0);
            }
            v16bf Ahi = __builtin_bit_cast(v16bf,
                (v8u){ah[0], ah[1], ah[2], ah[3], ah[4], ah[5], ah[6], ah[7]});
            v16bf Alo = __builtin_bit_cast(v16bf,
                (v8u){al[0], al[1], al[2], al[3], al[4], al[5], al[6], al[7]});

            const unsigned* bhiK = bhiF + (size_t)kk * 16 * 256;
            const unsigned* bloK = bloF + (size_t)kk * 16 * 256;
#pragma unroll
            for (int nn = 0; nn < 16; ++nn) {
                v8u rh = *(const v8u*)(bhiK + (size_t)nn * 256 + lane * 8);
                v8u rl = *(const v8u*)(bloK + (size_t)nn * 256 + lane * 8);
                v16bf Bh = __builtin_bit_cast(v16bf, rh);
                v16bf Bl = __builtin_bit_cast(v16bf, rl);
                // bf16x3 emulation of f32 GEMM: hi*hi + lo*hi + hi*lo
                acc[nn] = __builtin_amdgcn_wmma_f32_16x16x32_bf16(
                    false, Ahi, false, Bh, (short)0, acc[nn], false, false);
                acc[nn] = __builtin_amdgcn_wmma_f32_16x16x32_bf16(
                    false, Alo, false, Bh, (short)0, acc[nn], false, false);
                acc[nn] = __builtin_amdgcn_wmma_f32_16x16x32_bf16(
                    false, Ahi, false, Bl, (short)0, acc[nn], false, false);
            }
        }

        // ---------------- epilogue: gate + residual, row stats -------------
        const float* b2f = b2 + f * U_DIM;
        const float* Wgf = Wg + f * U_DIM;
        const float* bgf = bg + f * U_DIM;

        float s[8], s2[8];
#pragma unroll
        for (int j = 0; j < 8; ++j) { s[j] = 0.f; s2[j] = 0.f; }

#pragma unroll
        for (int nn = 0; nn < 16; ++nn) {
            int n = nn * 16 + l15;
            float b2v = b2f[n], wgv = Wgf[n], bgv = bgf[n];
#pragma unroll
            for (int j = 0; j < 8; ++j) {
                float g = sigmoidf(fmaf(xv[j], wgv, bgv));
                float r = fmaf(g, acc[nn][j] + b2v, xv[j]);
                acc[nn][j] = r;
                s[j] += r;
                s2[j] += r * r;
            }
        }
        // butterfly reduce across the 16-lane half (rows live per-half)
#pragma unroll
        for (int j = 0; j < 8; ++j) {
#pragma unroll
            for (int m = 1; m <= 8; m <<= 1) {
                s[j]  += __shfl_xor(s[j],  m, 32);
                s2[j] += __shfl_xor(s2[j], m, 32);
            }
        }
        float mu[8], rs[8];
#pragma unroll
        for (int j = 0; j < 8; ++j) {
            mu[j] = s[j] * (1.0f / 256.0f);
            float var = fmaf(-mu[j], mu[j], s2[j] * (1.0f / 256.0f));
            rs[j] = rsqrtf(var + 1e-3f);
        }

        // ---------------- LayerNorm, stash ln (fp16) + its mean ------------
        const float* gmf = gma + f * U_DIM;
        const float* btf = bta + f * U_DIM;
        float sl[8];
#pragma unroll
        for (int j = 0; j < 8; ++j) sl[j] = 0.f;

#pragma unroll
        for (int nn = 0; nn < 16; ++nn) {
            int n = nn * 16 + l15;
            float gv = gmf[n], bv = btf[n];
#pragma unroll
            for (int j = 0; j < 8; ++j) {
                float lnv = fmaf((acc[nn][j] - mu[j]) * rs[j], gv, bv);
                sl[j] += lnv;
                int m = j + 8 * half;
                lnbuf[((m * F_DIM + f) << 8) + n] = (_Float16)lnv;
            }
        }
#pragma unroll
        for (int j = 0; j < 8; ++j) {
#pragma unroll
            for (int m = 1; m <= 8; m <<= 1) sl[j] += __shfl_xor(sl[j], m, 32);
        }
        if (l15 == 0) {
#pragma unroll
            for (int j = 0; j < 8; ++j)
                Mbuf[(j + 8 * half) * F_DIM + f] = sl[j] * (1.0f / 256.0f);
        }
    }

    __syncthreads();

    // ---------------- selection weights: w = sigmoid(M @ Ws + bs) ----------
    {
        int m = tid >> 5;
        int f = tid & 31;
        float a = bs[f];
#pragma unroll
        for (int fp = 0; fp < F_DIM; ++fp)
            a = fmaf(Mbuf[m * F_DIM + fp], Ws[fp * F_DIM + f], a);
        wbuf[m * F_DIM + f] = sigmoidf(a);
    }
    __syncthreads();

    // ---------------- out[b,t,u] = sum_f w[f] * ln[f,u] --------------------
    {
        int m = tid >> 5;
        int u0 = tid & 31;
#pragma unroll
        for (int rep = 0; rep < 8; ++rep) {
            int u = u0 + rep * 32;
            float a = 0.f;
#pragma unroll
            for (int f2 = 0; f2 < F_DIM; ++f2)
                a = fmaf(wbuf[m * F_DIM + f2],
                         (float)lnbuf[((m * F_DIM + f2) << 8) + u], a);
            out[(size_t)(tok0 + m) * U_DIM + u] = a;
        }
    }
}

extern "C" void kernel_launch(void* const* d_in, const int* in_sizes, int n_in,
                              void* d_out, int out_size, void* d_ws, size_t ws_size,
                              hipStream_t stream) {
    const float* x   = (const float*)d_in[0];
    const float* W1  = (const float*)d_in[1];
    const float* b1  = (const float*)d_in[2];
    const float* W2  = (const float*)d_in[3];
    const float* b2  = (const float*)d_in[4];
    const float* Wg  = (const float*)d_in[5];
    const float* bg  = (const float*)d_in[6];
    const float* gma = (const float*)d_in[7];
    const float* bta = (const float*)d_in[8];
    const float* Ws  = (const float*)d_in[9];
    const float* bs  = (const float*)d_in[10];

    unsigned* bhi = (unsigned*)d_ws;                 // 4 MB bf16-hi B tiles
    unsigned* blo = bhi + WS_HI_DWORDS;              // 4 MB bf16-lo B tiles
    float* out = (float*)d_out;

    vsn_prep_w2<<<NBLOB, 256, 0, stream>>>(W2, bhi, blo);

    const size_t lds_bytes = (size_t)TOKS_PER_WG * F_DIM * U_DIM * 2   // ln fp16
                           + (size_t)TOKS_PER_WG * F_DIM * 4 * 2;      // M + w
    vsn_main<<<NTOK / TOKS_PER_WG, 512, lds_bytes, stream>>>(
        x, W1, b1, b2, Wg, bg, gma, bta, Ws, bs, bhi, blo, out);
}
